// DirichletEvidenceHead_58428735095480
// MI455X (gfx1250) — compile-verified, matrix operand order checked by reading, not run on hardware
//
#include <hip/hip_runtime.h>
#include <hip/hip_bf16.h>

// ---------------- problem constants ----------------
#define DDIM 4096
#define PDIM 256
#define KOUT 2
#define NTOK 16384          // B*T = 4*4096
#define EPS  1e-5f

// ---------------- tiling ----------------
#define MT      64                 // tokens per block (4 WMMA M-tiles)
#define KC      256                // k-chunk (global fp32 -> LDS bf16)
#define NCHUNK  (DDIM / KC)        // 16
#define WAVES   16                 // wave32
#define THREADS (WAVES * 32)       // 512

typedef __attribute__((ext_vector_type(16))) __bf16 v16bf;
typedef __attribute__((ext_vector_type(8)))  float  v8f;

union FragU { unsigned int u[8]; v16bf v; };   // 16 bf16 == 8 dwords
union AccU  { v8f v; float f[8]; };

__device__ __forceinline__ unsigned short f2bf(float f) {
    unsigned int u = __float_as_uint(f);
    return (unsigned short)((u + 0x7FFFu + ((u >> 16) & 1u)) >> 16); // RNE
}
__device__ __forceinline__ unsigned int pk2bf(float lo, float hi) {
    return (unsigned int)f2bf(lo) | ((unsigned int)f2bf(hi) << 16);
}

// gfx1250 async copy: each lane moves 16B global -> LDS, tracked by ASYNCcnt.
// Truncating a generic LDS pointer to 32 bits yields the LDS-relative offset
// (ISA: LDS aperture address = addr[31:0]).
__device__ __forceinline__ void async_lds_b128(void* lds, const void* gl) {
    unsigned loff = (unsigned)(size_t)lds;
    asm volatile("global_load_async_to_lds_b128 %0, %1, off"
                 :: "v"(loff), "v"(gl) : "memory");
}
__device__ __forceinline__ void wait_asynccnt0() {
    asm volatile("s_wait_asynccnt 0x0" ::: "memory");
}

// ---------------------------------------------------------------------------
// Prep: Wb[p][d] = bf16(ln_w[d] * W_proj[p][d]);  c1[p] = sum(ln_w*W);
//       c0[p] = sum(ln_b*W).   Wb (2 MB) stays resident in 192MB L2.
// ---------------------------------------------------------------------------
__global__ __launch_bounds__(256) void prep_weights(
    const float* __restrict__ Wp, const float* __restrict__ lnw,
    const float* __restrict__ lnb, unsigned short* __restrict__ Wb,
    float* __restrict__ c1, float* __restrict__ c0)
{
    __shared__ float r1[256], r0[256];
    const int p = blockIdx.x, tid = threadIdx.x;
    float a1 = 0.f, a0 = 0.f;
    for (int d = tid; d < DDIM; d += 256) {
        float w  = Wp[(size_t)p * DDIM + d];
        float ws = lnw[d] * w;
        a1 += ws;
        a0 += lnb[d] * w;
        Wb[(size_t)p * DDIM + d] = f2bf(ws);
    }
    r1[tid] = a1; r0[tid] = a0;
    __syncthreads();
    for (int s = 128; s > 0; s >>= 1) {
        if (tid < s) { r1[tid] += r1[tid + s]; r0[tid] += r0[tid + s]; }
        __syncthreads();
    }
    if (tid == 0) { c1[p] = r1[0]; c0[p] = r0[0]; }
}

// ---------------------------------------------------------------------------
// Main fused kernel: stream h once (HBM), LN-stats on the fly, bf16 WMMA
// GEMM against L2-resident weights, GELU + evidence head in epilogue.
// Double-buffered LDS A-tile: one workgroup barrier per k-chunk.
// ---------------------------------------------------------------------------
__global__ __launch_bounds__(THREADS) void fused_dirichlet_head(
    const float* __restrict__ h, const float* __restrict__ mask,
    const unsigned short* __restrict__ Wb, const float* __restrict__ c1,
    const float* __restrict__ c0, const float* __restrict__ Wev,
    const float* __restrict__ bev, float* __restrict__ out)
{
    __shared__ unsigned short hbf[2][MT][KC];     // bf16 A chunk, 2x32 KB
    __shared__ float psum[THREADS], psumsq[THREADS];
    __shared__ float smu[MT], srs[MT];
    __shared__ float gproj[MT][PDIM];             // GELU(proj) tile (64 KB)
    __shared__ float esh[MT][KOUT];
    __shared__ __align__(16) float cbuf[1024];    // [c1(256) c0(256) Wev(512)]

    const int tid  = threadIdx.x;
    const int lane = tid & 31;
    const int wave = tid >> 5;
    const int row0 = blockIdx.x * MT;

    // ---- kick off async const fetch (overlaps whole GEMM loop) ----
    if (wave == 0) {
        #pragma unroll
        for (int j = 0; j < 4; ++j) {
            async_lds_b128(&cbuf[j * 128 + lane * 4],       c1  + j * 128 + lane * 4);
            async_lds_b128(&cbuf[512 + j * 128 + lane * 4], Wev + j * 128 + lane * 4);
        }
    }

    // wave -> C-tile mapping: 1 M-tile x 4 N-tiles per wave
    const int mgroup = wave & 3;                  // M-tile 0..3
    const int ngroup = wave >> 2;                 // N-group 0..3 (4 tiles each)

    // cooperative-load mapping: 8 threads per token row, 32 cols each
    const int lrow = tid >> 3;                    // 0..63
    const int lsub = tid & 7;                     // 0..7
    const float* hrow = h + (size_t)(row0 + lrow) * DDIM + lsub * 32;

    // WMMA fragment geometry
    const int rowA = mgroup * 16 + (lane & 15);
    const int koff = (lane < 16) ? 0 : 8;
    const int colB = (lane & 15);

    float s1 = 0.f, s2 = 0.f;
    v8f acc[4] = {v8f{}, v8f{}, v8f{}, v8f{}};

    float4 stage[8];
    #pragma unroll
    for (int j = 0; j < 8; ++j)
        stage[j] = *(const float4*)(hrow + j * 4);

    // prologue: chunk 0 -> LDS buffer 0
    {
        unsigned short* dst = &hbf[0][lrow][lsub * 32];
        #pragma unroll
        for (int j = 0; j < 8; ++j) {
            float4 v = stage[j];
            s1 += v.x + v.y + v.z + v.w;
            s2 += v.x * v.x + v.y * v.y + v.z * v.z + v.w * v.w;
            *(uint2*)(dst + j * 4) = make_uint2(pk2bf(v.x, v.y), pk2bf(v.z, v.w));
        }
    }
    __syncthreads();

    for (int c = 0; c < NCHUNK; ++c) {
        const int buf = c & 1;

        // ---- issue next chunk's global loads early (hide behind WMMA) ----
        if (c + 1 < NCHUNK) {
            const float* hp = hrow + (c + 1) * KC;
            #pragma unroll
            for (int j = 0; j < 8; ++j)
                stage[j] = *(const float4*)(hp + j * 4);
            // prefetch next weight chunk (L2 temporal) -> global_prefetch_b8
            __builtin_prefetch(Wb + (size_t)(ngroup * 64 + colB) * DDIM + (c + 1) * KC, 0, 1);
        }

        // ---- WMMA over this chunk: 8 k-steps x 4 column tiles per wave ----
        const int kbaseG = c * KC;
        #pragma unroll
        for (int kk = 0; kk < KC / 32; ++kk) {
            FragU a;
            const unsigned short* ap = &hbf[buf][rowA][kk * 32 + koff];
            uint4 alo = *(const uint4*)ap;
            uint4 ahi = *(const uint4*)(ap + 16);
            a.u[0] = alo.x; a.u[1] = alo.y; a.u[2] = alo.z; a.u[3] = alo.w;
            a.u[4] = ahi.x; a.u[5] = ahi.y; a.u[6] = ahi.z; a.u[7] = ahi.w;

            #pragma unroll
            for (int t = 0; t < 4; ++t) {
                const int col = (ngroup * 4 + t) * 16 + colB;
                const unsigned short* bp =
                    Wb + (size_t)col * DDIM + kbaseG + kk * 32 + koff;
                uint4 blo = *(const uint4*)bp;
                uint4 bhi = *(const uint4*)(bp + 16);
                FragU b;
                b.u[0] = blo.x; b.u[1] = blo.y; b.u[2] = blo.z; b.u[3] = blo.w;
                b.u[4] = bhi.x; b.u[5] = bhi.y; b.u[6] = bhi.z; b.u[7] = bhi.w;

                acc[t] = __builtin_amdgcn_wmma_f32_16x16x32_bf16(
                    false, a.v, false, b.v, (short)0, acc[t], false, false);
            }
        }

        // ---- convert staged fp32 -> other LDS buffer, accumulate stats ----
        if (c + 1 < NCHUNK) {
            unsigned short* dst = &hbf[buf ^ 1][lrow][lsub * 32];
            #pragma unroll
            for (int j = 0; j < 8; ++j) {
                float4 v = stage[j];
                s1 += v.x + v.y + v.z + v.w;
                s2 += v.x * v.x + v.y * v.y + v.z * v.z + v.w * v.w;
                *(uint2*)(dst + j * 4) = make_uint2(pk2bf(v.x, v.y), pk2bf(v.z, v.w));
            }
        }
        __syncthreads();   // one barrier per chunk
    }

    // ---- retire async const fetch before epilogue reads cbuf ----
    if (wave == 0) wait_asynccnt0();

    // ---- finalize LayerNorm stats ----
    psum[tid] = s1; psumsq[tid] = s2;
    __syncthreads();
    if (tid < MT) {
        float a = 0.f, b = 0.f;
        #pragma unroll
        for (int j = 0; j < 8; ++j) { a += psum[tid * 8 + j]; b += psumsq[tid * 8 + j]; }
        float mu  = a * (1.f / DDIM);
        float var = b * (1.f / DDIM) - mu * mu;
        smu[tid] = mu;
        srs[tid] = rsqrtf(var + EPS);
    }
    __syncthreads();

    // ---- epilogue: proj = s*(acc - mu*c1) + c0 ; exact GELU -> LDS ----
    AccU au[4];
    #pragma unroll
    for (int t = 0; t < 4; ++t) au[t].v = acc[t];
    const int mbase = mgroup * 16 + ((lane >= 16) ? 8 : 0);
    #pragma unroll
    for (int t = 0; t < 4; ++t) {
        const int col = (ngroup * 4 + t) * 16 + colB;
        const float c1v = cbuf[col], c0v = cbuf[256 + col];
        #pragma unroll
        for (int i = 0; i < 8; ++i) {
            const int m = mbase + i;
            float proj = srs[m] * (au[t].f[i] - smu[m] * c1v) + c0v;
            float g = 0.5f * proj * (1.f + erff(proj * 0.70710678118654752f));
            gproj[m][col] = g;
        }
    }
    __syncthreads();

    // ---- evidence head: e = softplus(gproj @ Wev^T + bev) ----
    float* outU = out;                      // uncertainty [NTOK]
    float* outA = out + NTOK;               // alpha       [NTOK*KOUT]
    float* outE = out + NTOK + NTOK * KOUT; // evidence    [NTOK*KOUT]
    if (tid < MT * KOUT) {
        const int n = tid >> 1, k = tid & 1;
        float dot = bev[k];
        #pragma unroll 8
        for (int p = 0; p < PDIM; ++p)
            dot += gproj[n][p] * cbuf[512 + k * PDIM + p];
        float e = (dot > 20.f) ? dot : log1pf(expf(dot));
        esh[n][k] = e;
        const int gn = row0 + n;
        outA[gn * KOUT + k] = e + 1.f;
        outE[gn * KOUT + k] = e;
    }
    __syncthreads();
    if (tid < MT) {
        const int gn = row0 + tid;
        float ssum = esh[tid][0] + esh[tid][1] + 2.f;
        outU[gn] = ((float)KOUT / ssum) * mask[gn];
    }
}

// ---------------------------------------------------------------------------
extern "C" void kernel_launch(void* const* d_in, const int* in_sizes, int n_in,
                              void* d_out, int out_size, void* d_ws, size_t ws_size,
                              hipStream_t stream) {
    const float* h    = (const float*)d_in[0];   // [4,4096,4096]
    const float* mask = (const float*)d_in[1];   // [4,4096]
    const float* lnw  = (const float*)d_in[2];   // [4096]
    const float* lnb  = (const float*)d_in[3];   // [4096]
    const float* Wp   = (const float*)d_in[4];   // [256,4096]
    const float* Wev  = (const float*)d_in[5];   // [2,256]
    const float* bev  = (const float*)d_in[6];   // [2]

    unsigned short* Wb = (unsigned short*)d_ws;                       // 2 MB bf16
    float* c1 = (float*)((char*)d_ws + (size_t)PDIM * DDIM * 2);      // 1 KB
    float* c0 = c1 + PDIM;                                            // contiguous

    prep_weights<<<PDIM, 256, 0, stream>>>(Wp, lnw, lnb, Wb, c1, c0);
    fused_dirichlet_head<<<NTOK / MT, THREADS, 0, stream>>>(
        h, mask, Wb, c1, c0, Wev, bev, (float*)d_out);
}
